// SA_MambaLayer_70944269795622
// MI455X (gfx1250) — matrix operand order repaired and checked by reading, not compiled, not run on hardware
//
#include <hip/hip_runtime.h>
#include <hip/hip_bf16.h>

// ---------------------------------------------------------------------------
// Mamba layer for MI455X (gfx1250, wave32, WMMA, async-to-LDS).
// ---------------------------------------------------------------------------

typedef __bf16 bf16_t;
typedef __attribute__((ext_vector_type(16))) __bf16 v16bf;
typedef __attribute__((ext_vector_type(8)))  float  v8f;

#define DIM      256
#define D_STATE  64
#define D_CONV   4
#define D_INNER  512
#define DT_RANK  16
#define B_SZ     8
#define SEQ      4096
#define M_TOK    (B_SZ * SEQ)          // 32768 token rows
#define LN_EPS   1e-5f

// ---------------------------------------------------------------------------
// fp32 -> bf16 conversion (weights)
// ---------------------------------------------------------------------------
__global__ void cvt_f32_bf16(const float* __restrict__ in, bf16_t* __restrict__ out, int n) {
  int i = blockIdx.x * 256 + threadIdx.x;
  if (i < n) out[i] = (bf16_t)in[i];
}

// ---------------------------------------------------------------------------
// LayerNorm over DIM=256, one token per 256-thread block, bf16 output
// ---------------------------------------------------------------------------
__global__ void ln_kernel(const float* __restrict__ x,
                          const float* __restrict__ lnw,
                          const float* __restrict__ lnb,
                          bf16_t* __restrict__ xn) {
  const int m = blockIdx.x;
  const int t = threadIdx.x;
  float v = x[(size_t)m * DIM + t];
  float s = v, q = v * v;
  #pragma unroll
  for (int o = 16; o > 0; o >>= 1) {
    s += __shfl_down(s, o, 32);
    q += __shfl_down(q, o, 32);
  }
  __shared__ float rs[8], rq[8];
  if ((t & 31) == 0) { rs[t >> 5] = s; rq[t >> 5] = q; }
  __syncthreads();
  if (t == 0) {
    float S = 0.f, Q = 0.f;
    #pragma unroll
    for (int i = 0; i < 8; ++i) { S += rs[i]; Q += rq[i]; }
    rs[0] = S * (1.f / DIM);
    rq[0] = Q * (1.f / DIM);
  }
  __syncthreads();
  const float mu  = rs[0];
  const float var = rq[0] - mu * mu;
  const float inv = rsqrtf(var + LN_EPS);
  xn[(size_t)m * DIM + t] = (bf16_t)((v - mu) * inv * lnw[t] + lnb[t]);
}

// ---------------------------------------------------------------------------
// WMMA bf16 GEMM:  C[M,N] = A[M,K] * W[N,K]^T
// Wave tile: 64(M) x 16(N) -> 4 accumulators; B fragment loaded once per
// K-chunk and reused across the 4 M-subtiles (2.5 b128 loads per v_wmma).
// Fragment layouts per CDNA5 ISA 7.12.2:
//   A frag (16x32 bf16):  lane<16: m=lane,    K = {k0..k0+7, k0+16..k0+23}
//                         lane>=16: m=lane-16, K = {k0+8..k0+15, k0+24..k0+31}
//   B frag (32x16 bf16):  n = lane&15, K = same split as A frag
//   D (f32, 8 VGPRs):     m = j + 8*(lane>=16), n = lane&15
// MODE 0: in_proj  -> C0[m*1024 + n]
// MODE 1: x_proj   -> n<16: C0(dt)[m*16+n]; n<80: C1(B)[m*64+n-16]; else C2(Cssm)[m*64+n-80]
// MODE 2: out_proj -> C0[m*256 + n]
// ---------------------------------------------------------------------------
template <int MODE>
__global__ void wmma_gemm(const bf16_t* __restrict__ A,
                          const bf16_t* __restrict__ W,
                          float* __restrict__ C0,
                          float* __restrict__ C1,
                          float* __restrict__ C2,
                          int Mtiles64, int Ntiles, int K, int lda, int ldw) {
  const int wave = threadIdx.x >> 5;
  const int lane = threadIdx.x & 31;
  const int tile = blockIdx.x * 8 + wave;
  if (tile >= Mtiles64 * Ntiles) return;
  const int tm = tile / Ntiles;          // 64-row block index
  const int tn = tile % Ntiles;

  const int half  = lane >> 4;           // K-half select
  const int mrowA = tm * 64 + (lane & 15);
  const int ncolB = tn * 16 + (lane & 15);

  union FragU { uint4 u[2]; v16bf v; };
  v8f acc0 = {}, acc1 = {}, acc2 = {}, acc3 = {};

  const bf16_t* aBase = A + (size_t)mrowA * lda + half * 8;
  const bf16_t* bBase = W + (size_t)ncolB * ldw + half * 8;
  const size_t aStep = (size_t)16 * lda;           // 16 rows per M-subtile

  for (int k0 = 0; k0 < K; k0 += 32) {
    FragU fb;
    fb.u[0] = *(const uint4*)(bBase + k0);
    fb.u[1] = *(const uint4*)(bBase + k0 + 16);

    FragU fa;
    fa.u[0] = *(const uint4*)(aBase + k0);
    fa.u[1] = *(const uint4*)(aBase + k0 + 16);
    acc0 = __builtin_amdgcn_wmma_f32_16x16x32_bf16(false, fa.v, false, fb.v, (short)0, acc0, false, false);

    fa.u[0] = *(const uint4*)(aBase + aStep + k0);
    fa.u[1] = *(const uint4*)(aBase + aStep + k0 + 16);
    acc1 = __builtin_amdgcn_wmma_f32_16x16x32_bf16(false, fa.v, false, fb.v, (short)0, acc1, false, false);

    fa.u[0] = *(const uint4*)(aBase + 2 * aStep + k0);
    fa.u[1] = *(const uint4*)(aBase + 2 * aStep + k0 + 16);
    acc2 = __builtin_amdgcn_wmma_f32_16x16x32_bf16(false, fa.v, false, fb.v, (short)0, acc2, false, false);

    fa.u[0] = *(const uint4*)(aBase + 3 * aStep + k0);
    fa.u[1] = *(const uint4*)(aBase + 3 * aStep + k0 + 16);
    acc3 = __builtin_amdgcn_wmma_f32_16x16x32_bf16(false, fa.v, false, fb.v, (short)0, acc3, false, false);
  }

  const int n = tn * 16 + (lane & 15);
  v8f accs[4] = {acc0, acc1, acc2, acc3};
  #pragma unroll
  for (int s = 0; s < 4; ++s) {
    #pragma unroll
    for (int j = 0; j < 8; ++j) {
      const int m = tm * 64 + s * 16 + j + half * 8;
      const float r = accs[s][j];
      if constexpr (MODE == 0) {
        C0[(size_t)m * 1024 + n] = r;
      } else if constexpr (MODE == 1) {
        if (n < DT_RANK)                C0[(size_t)m * DT_RANK + n] = r;
        else if (n < DT_RANK + D_STATE) C1[(size_t)m * D_STATE + (n - DT_RANK)] = r;
        else                            C2[(size_t)m * D_STATE + (n - DT_RANK - D_STATE)] = r;
      } else {
        C0[(size_t)m * DIM + n] = r;
      }
    }
  }
}

// ---------------------------------------------------------------------------
// Causal depthwise conv (D_CONV=4) + bias + SiLU.  xz[M,1024] cols 0..511 -> xs
// ---------------------------------------------------------------------------
__global__ void conv_silu(const float* __restrict__ xz,
                          const float* __restrict__ cw,
                          const float* __restrict__ cb,
                          float* __restrict__ xsf,
                          bf16_t* __restrict__ xsb) {
  const size_t i = (size_t)blockIdx.x * 256 + threadIdx.x;   // over M*512
  const int d = (int)(i & (D_INNER - 1));
  const size_t m = i >> 9;
  const int l = (int)(m & (SEQ - 1));
  const size_t rowb = m - (size_t)l;                         // batch row base
  float acc = cb[d];
  #pragma unroll
  for (int k = 0; k < D_CONV; ++k) {
    const int ls = l - (D_CONV - 1) + k;
    if (ls >= 0)
      acc += xz[(rowb + (size_t)ls) * 1024 + d] * cw[d * D_CONV + k];
  }
  acc = acc / (1.f + __expf(-acc));                          // SiLU
  xsf[i] = acc;
  xsb[i] = (bf16_t)acc;
}

// ---------------------------------------------------------------------------
// delta = softplus(dt[M,16] * dt_proj_w[512,16]^T + dt_proj_b)
// ---------------------------------------------------------------------------
__global__ void dtproj_softplus(const float* __restrict__ dtraw,
                                const float* __restrict__ dtw,
                                const float* __restrict__ dtb,
                                float* __restrict__ delta) {
  const size_t i = (size_t)blockIdx.x * 256 + threadIdx.x;   // over M*512
  const int n = (int)(i & (D_INNER - 1));
  const size_t m = i >> 9;
  float acc = dtb[n];
  #pragma unroll
  for (int r = 0; r < DT_RANK; ++r)
    acc += dtraw[m * DT_RANK + r] * dtw[n * DT_RANK + r];
  delta[i] = (acc > 20.f) ? acc : log1pf(__expf(acc));
}

// ---------------------------------------------------------------------------
// Selective-scan. Block = 128 threads covers 16 channels (8 lanes/channel,
// 8 states/lane in VGPRs). Per-step B/C staged into LDS with CDNA5
// global_load_async_to_lds_b32 (ASYNCcnt), fenced by s_wait_asynccnt before
// the workgroup barrier. Epilogue fuses D-skip and SiLU(z) gate -> bf16.
// ---------------------------------------------------------------------------
__global__ void ssm_scan(const float* __restrict__ delta,
                         const float* __restrict__ xsf,
                         const float* __restrict__ Bb,
                         const float* __restrict__ Cb,
                         const float* __restrict__ A_log,
                         const float* __restrict__ Dsk,
                         const float* __restrict__ xz,
                         bf16_t* __restrict__ yb) {
  const int b     = blockIdx.x >> 5;             // 32 blocks per batch
  const int dbase = (blockIdx.x & 31) * 16;
  const int t     = threadIdx.x;                 // 0..127
  const int chl   = t >> 3;                      // local channel 0..15
  const int ch    = dbase + chl;
  const int sub   = t & 7;
  const int s0    = sub * 8;

  float A2[8], h[8];
  #pragma unroll
  for (int i = 0; i < 8; ++i) {
    A2[i] = -__expf(A_log[(size_t)ch * D_STATE + s0 + i]) * 1.44269504f; // * log2(e)
    h[i]  = 0.f;
  }
  const float Dv = Dsk[ch];

  __shared__ float sB[D_STATE], sC[D_STATE], sD[16], sX[16];
  const size_t mbase = (size_t)b * SEQ;

  // Async-to-LDS staging: threads 0..63 own sB, threads 64..127 own sC.
  // LDS byte offset = low 32 bits of the generic shared address (ISA 10.2).
  const unsigned ldsOff = (t < 64) ? (unsigned)(unsigned long long)&sB[t]
                                   : (unsigned)(unsigned long long)&sC[t - 64];
  const float* gsrc = (t < 64) ? (Bb + mbase * D_STATE + t)
                               : (Cb + mbase * D_STATE + (t - 64));

  for (int l = 0; l < SEQ; ++l) {
    const size_t m = mbase + l;
    // one 32-bit async DMA element per thread: global -> LDS
    asm volatile("global_load_async_to_lds_b32 %0, %1, off"
                 :: "v"(ldsOff), "v"(gsrc) : "memory");
    gsrc += D_STATE;
    if (t < 16)      sD[t]      = delta[m * D_INNER + dbase + t];
    else if (t < 32) sX[t - 16] = xsf[m * D_INNER + dbase + (t - 16)];
    asm volatile("s_wait_asynccnt 0x0" ::: "memory");
    __syncthreads();

    const float dl = sD[chl];
    const float xv = sX[chl];
    const float dx = dl * xv;
    float y = 0.f;
    #pragma unroll
    for (int i = 0; i < 8; ++i) {
      const float da = exp2f(dl * A2[i]);        // native v_exp pipe
      h[i] = da * h[i] + dx * sB[s0 + i];
      y += h[i] * sC[s0 + i];
    }
    y += __shfl_xor(y, 1, 32);
    y += __shfl_xor(y, 2, 32);
    y += __shfl_xor(y, 4, 32);
    if (sub == 0) {
      const float z = xz[m * 1024 + D_INNER + ch];
      const float g = z / (1.f + __expf(-z));    // SiLU gate
      yb[m * D_INNER + ch] = (bf16_t)((y + xv * Dv) * g);
    }
    __syncthreads();
  }
}

// ---------------------------------------------------------------------------
// Host launcher
// ---------------------------------------------------------------------------
extern "C" void kernel_launch(void* const* d_in, const int* in_sizes, int n_in,
                              void* d_out, int out_size, void* d_ws, size_t ws_size,
                              hipStream_t stream) {
  const float* x        = (const float*)d_in[0];
  const float* ln_w     = (const float*)d_in[1];
  const float* ln_b     = (const float*)d_in[2];
  const float* in_pw    = (const float*)d_in[3];   // [1024, 256]
  const float* conv_w   = (const float*)d_in[4];   // [512, 1, 4]
  const float* conv_b   = (const float*)d_in[5];
  const float* x_pw     = (const float*)d_in[6];   // [144, 512]
  const float* dt_pw    = (const float*)d_in[7];   // [512, 16]
  const float* dt_pb    = (const float*)d_in[8];
  const float* A_log    = (const float*)d_in[9];   // [512, 64]
  const float* D_skip   = (const float*)d_in[10];
  const float* out_pw   = (const float*)d_in[11];  // [256, 512]

  float* outp = (float*)d_out;                 // [M, 256]
  float* Cout = outp + (size_t)M_TOK * DIM;    // [M, 64] (second output)

  // ---- workspace layout (256 B aligned blocks) ----
  char* w = (char*)d_ws;
  auto take = [&](size_t bytes) { char* p = w; w += (bytes + 255) & ~(size_t)255; return p; };
  bf16_t* xn    = (bf16_t*)take((size_t)M_TOK * DIM * 2);
  float*  xz    = (float*) take((size_t)M_TOK * 1024 * 4);
  float*  xsf   = (float*) take((size_t)M_TOK * D_INNER * 4);
  bf16_t* xsb   = (bf16_t*)take((size_t)M_TOK * D_INNER * 2);
  float*  dtraw = (float*) take((size_t)M_TOK * DT_RANK * 4);
  float*  Bbuf  = (float*) take((size_t)M_TOK * D_STATE * 4);
  float*  dbuf  = (float*) take((size_t)M_TOK * D_INNER * 4);
  bf16_t* ybuf  = (bf16_t*)take((size_t)M_TOK * D_INNER * 2);
  bf16_t* wip   = (bf16_t*)take((size_t)1024 * DIM * 2);
  bf16_t* wxp   = (bf16_t*)take((size_t)(DT_RANK + 2 * D_STATE) * D_INNER * 2);
  bf16_t* wop   = (bf16_t*)take((size_t)DIM * D_INNER * 2);

  // ---- weight conversion to bf16 ----
  {
    int n1 = 1024 * DIM, n2 = (DT_RANK + 2 * D_STATE) * D_INNER, n3 = DIM * D_INNER;
    cvt_f32_bf16<<<(n1 + 255) / 256, 256, 0, stream>>>(in_pw, wip, n1);
    cvt_f32_bf16<<<(n2 + 255) / 256, 256, 0, stream>>>(x_pw, wxp, n2);
    cvt_f32_bf16<<<(n3 + 255) / 256, 256, 0, stream>>>(out_pw, wop, n3);
  }

  // ---- LayerNorm ----
  ln_kernel<<<M_TOK, 256, 0, stream>>>(x, ln_w, ln_b, xn);

  // ---- in_proj: [M,256] x [1024,256]^T -> xz[M,1024] ----
  {
    int mt = M_TOK / 64, nt = 1024 / 16;
    int blocks = (mt * nt + 7) / 8;
    wmma_gemm<0><<<blocks, 256, 0, stream>>>(xn, wip, xz, nullptr, nullptr,
                                             mt, nt, DIM, DIM, DIM);
  }

  // ---- causal depthwise conv + SiLU ----
  conv_silu<<<(M_TOK * D_INNER) / 256, 256, 0, stream>>>(xz, conv_w, conv_b, xsf, xsb);

  // ---- x_proj: [M,512] x [144,512]^T -> dt / B / C ----
  {
    int mt = M_TOK / 64, nt = (DT_RANK + 2 * D_STATE) / 16;   // 9 tiles
    int blocks = (mt * nt + 7) / 8;
    wmma_gemm<1><<<blocks, 256, 0, stream>>>(xsb, wxp, dtraw, Bbuf, Cout,
                                             mt, nt, D_INNER, D_INNER, D_INNER);
  }

  // ---- dt_proj + softplus ----
  dtproj_softplus<<<(M_TOK * D_INNER) / 256, 256, 0, stream>>>(dtraw, dt_pw, dt_pb, dbuf);

  // ---- selective scan (fused D-skip + SiLU gate, async-to-LDS staging) ----
  ssm_scan<<<B_SZ * 32, 128, 0, stream>>>(dbuf, xsf, Bbuf, Cout, A_log, D_skip, xz, ybuf);

  // ---- out_proj: [M,512] x [256,512]^T -> out[M,256] ----
  {
    int mt = M_TOK / 64, nt = DIM / 16;
    int blocks = (mt * nt + 7) / 8;
    wmma_gemm<2><<<blocks, 256, 0, stream>>>(ybuf, wop, outp, nullptr, nullptr,
                                             mt, nt, D_INNER, D_INNER, D_INNER);
  }
}
